// Attention_62989990363302
// MI455X (gfx1250) — matrix operand order repaired and checked by reading, not compiled, not run on hardware
//
#include <hip/hip_runtime.h>
#include <hip/hip_bf16.h>

#define BB 2
#define SS 2048
#define HH 12
#define DM 768
#define DH 64

typedef __bf16 bf16_t;
typedef __attribute__((ext_vector_type(16))) __bf16 v16bf;
typedef __attribute__((ext_vector_type(8)))  __bf16 v8bf;
typedef __attribute__((ext_vector_type(8)))  float  v8f;
typedef __attribute__((ext_vector_type(4)))  float  v4f;

__device__ __forceinline__ v8f wmma_bf16(v16bf a, v16bf b, v8f c) {
    return __builtin_amdgcn_wmma_f32_16x16x32_bf16(false, a, false, b, (short)0, c,
                                                   false, false);
}

// A-fragment (16x32 bf16, MxK): lane holds row (lane&15); elems 0..7 -> K = k0 + half*8 + i,
// elems 8..15 -> K = k0 + 16 + half*8 + i.  Row-major source, leading dim `ld` (elements).
__device__ __forceinline__ v16bf load_a_bf16(const bf16_t* base, int ld, int row0, int k0,
                                             int lane) {
    int r = lane & 15, hf = lane >> 4;
    const bf16_t* p = base + (size_t)(row0 + r) * ld + k0 + hf * 8;
    v8bf lo = *(const v8bf*)p;
    v8bf hi = *(const v8bf*)(p + 16);
    v16bf a;
#pragma unroll
    for (int i = 0; i < 8; i++) { a[i] = lo[i]; a[8 + i] = hi[i]; }
    return a;
}

// Same A-fragment but from an f32 row-major source (converted to bf16 in regs).
__device__ __forceinline__ v16bf load_a_f32(const float* base, int ld, int row0, int k0,
                                            int lane) {
    int r = lane & 15, hf = lane >> 4;
    const float* p = base + (size_t)(row0 + r) * ld + k0 + hf * 8;
    v4f f0 = *(const v4f*)p;
    v4f f1 = *(const v4f*)(p + 4);
    v4f f2 = *(const v4f*)(p + 16);
    v4f f3 = *(const v4f*)(p + 20);
    v16bf a;
#pragma unroll
    for (int i = 0; i < 4; i++) {
        a[i]      = (__bf16)f0[i];
        a[4 + i]  = (__bf16)f1[i];
        a[8 + i]  = (__bf16)f2[i];
        a[12 + i] = (__bf16)f3[i];
    }
    return a;
}

// B-fragment (32x16 bf16, KxN): lane holds column n0+(lane&15); elems 0..7 -> K = k0 + half*8 + i,
// elems 8..15 -> K = k0 + 16 + half*8 + i.  Source stored n-major / k-contiguous with leading dim `ld`.
__device__ __forceinline__ v16bf load_b_bf16(const bf16_t* base, int ld, int n0, int k0,
                                             int lane) {
    int n = lane & 15, hf = lane >> 4;
    const bf16_t* p = base + (size_t)(n0 + n) * ld + k0 + hf * 8;
    v8bf lo = *(const v8bf*)p;
    v8bf hi = *(const v8bf*)(p + 16);
    v16bf b;
#pragma unroll
    for (int i = 0; i < 8; i++) { b[i] = lo[i]; b[8 + i] = hi[i]; }
    return b;
}

// ---- Kernel 1: transpose weights to k-contiguous bf16 --------------------------------
__global__ void prep_weights(const float* __restrict__ WQ, const float* __restrict__ WK,
                             const float* __restrict__ WV, const float* __restrict__ WO,
                             bf16_t* __restrict__ wqt, bf16_t* __restrict__ wkt,
                             bf16_t* __restrict__ wvt, bf16_t* __restrict__ wot) {
    int idx = blockIdx.x * blockDim.x + threadIdx.x;
    const int N = HH * DM * DH;
    if (idx >= N) return;
    // W_[QKV]: [h][m][e] -> wt[h][e][m]
    int h = idx / (DM * DH);
    int rem = idx % (DM * DH);
    int m = rem / DH, e = rem % DH;
    size_t o = ((size_t)h * DH + e) * DM + m;
    wqt[o] = (__bf16)WQ[idx];
    wkt[o] = (__bf16)WK[idx];
    wvt[o] = (__bf16)WV[idx];
    // W_O: [h][e][m] -> wot[h][m][e]
    int h2 = idx / (DH * DM);
    int rem2 = idx % (DH * DM);
    int e2 = rem2 / DM, m2 = rem2 % DM;
    wot[((size_t)h2 * DM + m2) * DH + e2] = (__bf16)WO[idx];
}

// ---- Kernel 2: Q/K/V projection.  Each wave: 16 rows x 64 cols, K=768 in 24 WMMA steps.
__global__ __launch_bounds__(256) void proj_kernel(const float* __restrict__ X,
                                                   const bf16_t* __restrict__ WT,
                                                   const float* __restrict__ bias,
                                                   bf16_t* __restrict__ OUT,
                                                   int transposed) {
    int lane = threadIdx.x & 31;
    int wv = threadIdx.x >> 5;
    int bh = blockIdx.y;
    int b = bh / HH, h = bh % HH;
    int sbase = (blockIdx.x * 8 + wv) * 16;
    const float* Xbh = X + ((size_t)b * SS * HH + h) * DM;   // row stride = HH*DM
    const bf16_t* Wb = WT + (size_t)h * DH * DM;

    v8f acc[4] = {v8f{}, v8f{}, v8f{}, v8f{}};
    for (int kk = 0; kk < DM; kk += 32) {
        v16bf a = load_a_f32(Xbh, HH * DM, sbase, kk, lane);
#pragma unroll
        for (int nt = 0; nt < 4; nt++) {
            v16bf bfrag = load_b_bf16(Wb, DM, nt * 16, kk, lane);
            acc[nt] = wmma_bf16(a, bfrag, acc[nt]);
        }
    }
    int r0 = (lane >> 4) * 8, c0 = lane & 15;
#pragma unroll
    for (int nt = 0; nt < 4; nt++) {
        int col = nt * 16 + c0;
        float bv = bias[h * DH + col];
#pragma unroll
        for (int j = 0; j < 8; j++) {
            float val = acc[nt][j] + bv;
            int row = sbase + r0 + j;
            if (!transposed)
                OUT[((size_t)bh * SS + row) * DH + col] = (__bf16)val;   // [b,h,s,dh]
            else
                OUT[((size_t)bh * DH + col) * SS + row] = (__bf16)val;   // [b,h,dh,s]
        }
    }
}

// ---- Kernel 3: causal flash attention.  One wave = one 16-query tile. ---------------
__global__ __launch_bounds__(256) void attn_kernel(const bf16_t* __restrict__ Q,
                                                   const bf16_t* __restrict__ K,
                                                   const bf16_t* __restrict__ VT,
                                                   bf16_t* __restrict__ Z) {
    __shared__ __align__(16) bf16_t pbuf[8][16 * 32];   // per-wave P repack buffer
    int lane = threadIdx.x & 31;
    int wv = threadIdx.x >> 5;
    int bh = blockIdx.y;
    int qb = (blockIdx.x * 8 + wv) * 16;
    const bf16_t* Qb = Q + (size_t)bh * SS * DH;
    const bf16_t* Kb = K + (size_t)bh * SS * DH;
    const bf16_t* Vb = VT + (size_t)bh * DH * SS;

    v16bf qf0 = load_a_bf16(Qb, DH, qb, 0, lane);
    v16bf qf1 = load_a_bf16(Qb, DH, qb, 32, lane);

    v8f o[4] = {v8f{}, v8f{}, v8f{}, v8f{}};
    float m[8], l[8];
#pragma unroll
    for (int j = 0; j < 8; j++) { m[j] = -3.0e38f; l[j] = 0.0f; }

    int r0 = (lane >> 4) * 8, c0 = lane & 15;
    int nkt = (qb + 47) >> 5;   // key tiles of 32 touching the causal region

    // prefetch first key tile (K rows + V^T rows) into near caches
    __builtin_prefetch(Kb + (size_t)lane * DH, 0, 0);                 // keys 0..31
    __builtin_prefetch(Vb + (size_t)lane * SS, 0, 0);                 // dh 0..31
    __builtin_prefetch(Vb + (size_t)(lane + 32) * SS, 0, 0);          // dh 32..63

    for (int kt = 0; kt < nkt; kt++) {
        int kb = kt * 32;

        // prefetch next key tile while this one is consumed
        if (kt + 1 < nkt) {
            int nk2 = kb + 32;
            __builtin_prefetch(Kb + (size_t)(nk2 + lane) * DH, 0, 0);
            __builtin_prefetch(Vb + (size_t)lane * SS + nk2, 0, 0);
            __builtin_prefetch(Vb + (size_t)(lane + 32) * SS + nk2, 0, 0);
        }

        // ---- scores: S = Q * K^T (two 16-key column groups, K-dim 64 in 2 chunks)
        v8f s0 = v8f{}, s1 = v8f{};
        s0 = wmma_bf16(qf0, load_b_bf16(Kb, DH, kb, 0, lane), s0);
        s0 = wmma_bf16(qf1, load_b_bf16(Kb, DH, kb, 32, lane), s0);
        s1 = wmma_bf16(qf0, load_b_bf16(Kb, DH, kb + 16, 0, lane), s1);
        s1 = wmma_bf16(qf1, load_b_bf16(Kb, DH, kb + 16, 32, lane), s1);

        bool partial = (kb + 31 > qb);
#pragma unroll
        for (int j = 0; j < 8; j++) {
            s0[j] *= 0.125f;   // 1/sqrt(64)
            s1[j] *= 0.125f;
            if (partial) {
                int row = qb + r0 + j;
                if (kb + c0 > row)      s0[j] = -1.0e30f;
                if (kb + 16 + c0 > row) s1[j] = -1.0e30f;
            }
        }
        // ---- online softmax (row stats across the 16 lanes of each half-wave)
        float mn[8], alpha[8], rs[8];
#pragma unroll
        for (int j = 0; j < 8; j++) mn[j] = fmaxf(s0[j], s1[j]);
        for (int off = 1; off < 16; off <<= 1)
#pragma unroll
            for (int j = 0; j < 8; j++) mn[j] = fmaxf(mn[j], __shfl_xor(mn[j], off, 32));
#pragma unroll
        for (int j = 0; j < 8; j++) {
            float mnew = fmaxf(m[j], mn[j]);
            alpha[j] = __expf(m[j] - mnew);
            m[j] = mnew;
            s0[j] = __expf(s0[j] - mnew);
            s1[j] = __expf(s1[j] - mnew);
            rs[j] = s0[j] + s1[j];
        }
        for (int off = 1; off < 16; off <<= 1)
#pragma unroll
            for (int j = 0; j < 8; j++) rs[j] += __shfl_xor(rs[j], off, 32);
#pragma unroll
        for (int j = 0; j < 8; j++) l[j] = l[j] * alpha[j] + rs[j];
#pragma unroll
        for (int nt = 0; nt < 4; nt++)
#pragma unroll
            for (int j = 0; j < 8; j++) o[nt][j] *= alpha[j];

        // ---- repack P (C layout: lane=col, rows=half*8+j) -> A layout via wave-local LDS
        bf16_t* pb = pbuf[wv];
#pragma unroll
        for (int j = 0; j < 8; j++) {
            int row = r0 + j;
            pb[row * 32 + c0]      = (__bf16)s0[j];
            pb[row * 32 + 16 + c0] = (__bf16)s1[j];
        }
        asm volatile("s_wait_dscnt 0" ::: "memory");   // wave-local: no block barrier
        v16bf pf = load_a_bf16(pb, 32, 0, 0, lane);

        // ---- O += P * V  (V^T stored dh-major so B-frag reads are 16B contiguous runs)
#pragma unroll
        for (int nt = 0; nt < 4; nt++) {
            v16bf vf = load_b_bf16(Vb, SS, nt * 16, kb, lane);
            o[nt] = wmma_bf16(pf, vf, o[nt]);
        }
        asm volatile("" ::: "memory");
    }
    // ---- normalize and store z (bf16, [b,h,s,dh])
#pragma unroll
    for (int nt = 0; nt < 4; nt++) {
#pragma unroll
        for (int j = 0; j < 8; j++) {
            float val = o[nt][j] / l[j];
            Z[((size_t)bh * SS + qb + r0 + j) * DH + nt * 16 + c0] = (__bf16)val;
        }
    }
}

// ---- Kernel 4: per-head output projection z[S,64] x W_O[64,768] + b_O/H -------------
__global__ __launch_bounds__(256) void out_proj(const bf16_t* __restrict__ Z,
                                                const bf16_t* __restrict__ WOT,
                                                const float* __restrict__ bO,
                                                float* __restrict__ OUT) {
    int lane = threadIdx.x & 31;
    int wv = threadIdx.x >> 5;
    int bh = blockIdx.z;
    int b = bh / HH, h = bh % HH;
    int sbase = (blockIdx.x * 8 + wv) * 16;
    int nb = blockIdx.y * 64;
    const bf16_t* Zb = Z + (size_t)bh * SS * DH;
    const bf16_t* Wb = WOT + (size_t)h * DM * DH;

    v16bf a0 = load_a_bf16(Zb, DH, sbase, 0, lane);
    v16bf a1 = load_a_bf16(Zb, DH, sbase, 32, lane);
    int r0 = (lane >> 4) * 8, c0 = lane & 15;
#pragma unroll
    for (int nt = 0; nt < 4; nt++) {
        v8f acc = v8f{};
        acc = wmma_bf16(a0, load_b_bf16(Wb, DH, nb + nt * 16, 0, lane), acc);
        acc = wmma_bf16(a1, load_b_bf16(Wb, DH, nb + nt * 16, 32, lane), acc);
        int mcol = nb + nt * 16 + c0;
        float bias = bO[mcol] * (1.0f / HH);
#pragma unroll
        for (int j = 0; j < 8; j++) {
            int row = sbase + r0 + j;
            OUT[(((size_t)b * SS + row) * HH + h) * DM + mcol] = acc[j] + bias;
        }
    }
}

extern "C" void kernel_launch(void* const* d_in, const int* in_sizes, int n_in,
                              void* d_out, int out_size, void* d_ws, size_t ws_size,
                              hipStream_t stream) {
    const float* nq = (const float*)d_in[0];
    const float* nk = (const float*)d_in[1];
    const float* nv = (const float*)d_in[2];
    const float* WQ = (const float*)d_in[3];
    const float* bQ = (const float*)d_in[4];
    const float* WK = (const float*)d_in[5];
    const float* bK = (const float*)d_in[6];
    const float* WV = (const float*)d_in[7];
    const float* bV = (const float*)d_in[8];
    const float* WO = (const float*)d_in[9];
    const float* bO = (const float*)d_in[10];

    const size_t WT  = (size_t)HH * DM * DH;       // 589824 bf16
    const size_t QKV = (size_t)BB * HH * SS * DH;  // 3145728 bf16
    bf16_t* wqt = (bf16_t*)d_ws;
    bf16_t* wkt = wqt + WT;
    bf16_t* wvt = wkt + WT;
    bf16_t* wot = wvt + WT;
    bf16_t* qb  = wot + WT;
    bf16_t* kb  = qb + QKV;
    bf16_t* vtb = kb + QKV;
    bf16_t* zb  = vtb + QKV;

    prep_weights<<<(HH * DM * DH + 255) / 256, 256, 0, stream>>>(WQ, WK, WV, WO,
                                                                 wqt, wkt, wvt, wot);
    dim3 g(SS / 128, BB * HH);
    proj_kernel<<<g, 256, 0, stream>>>(nq, wqt, bQ, qb, 0);
    proj_kernel<<<g, 256, 0, stream>>>(nk, wkt, bK, kb, 0);
    proj_kernel<<<g, 256, 0, stream>>>(nv, wvt, bV, vtb, 1);
    attn_kernel<<<g, 256, 0, stream>>>(qb, kb, vtb, zb);
    dim3 go(SS / 128, DM / 64, BB * HH);
    out_proj<<<go, 256, 0, stream>>>(zb, wot, bO, (float*)d_out);
}
